// MultiHeadLatentAttention_83339545411939
// MI455X (gfx1250) — compile-verified, matrix operand order checked by reading, not run on hardware
//
#include <hip/hip_runtime.h>

// ---------------------------------------------------------------------------
// MLA-style attention pipeline for MI455X (gfx1250, wave32, WMMA).
// All GEMMs and both attention einsums run through v_wmma_f32_16x16x32_bf16.
// LDS staging of non-transposed tiles uses CDNA5 async global->LDS copies
// (GLOBAL_LOAD_ASYNC_TO_LDS_B128, ASYNCcnt).
// ---------------------------------------------------------------------------

typedef unsigned short u16;
typedef __attribute__((ext_vector_type(16))) __bf16 v16bf;
typedef __attribute__((ext_vector_type(8)))  float  v8f;

#define B_    2
#define T_    2048
#define D_    2048
#define NH_   16
#define HD_   128
#define R_    64
#define CQ_   1536
#define CKV_  512
#define DQK_  192           // HD + R
#define MROWS 4096          // B*T

__device__ __forceinline__ u16 f2bf(float f) {
  unsigned u = __float_as_uint(f);
  u += 0x7fffu + ((u >> 16) & 1u);   // round-to-nearest-even
  return (u16)(u >> 16);
}

union FragU { v16bf bf; uint4 q[2]; u16 s[16]; };

// Load a 16x32 (rows x K) bf16 fragment in the CDNA5 WMMA A/B layout:
// lane L (0-15) holds row L, K chunks {0..7, 16..23}; lane L+16 holds
// row L, K chunks {8..15, 24..31}.  Works for both A and (column-major
// staged) B operands.  p points at a row-major [16 x stride] tile.
__device__ __forceinline__ v16bf load_frag(const u16* p, int stride) {
  const int lane = threadIdx.x & 31;
  const u16* row = p + (lane & 15) * stride + (lane >> 4) * 8;
  FragU f;
  f.q[0] = *(const uint4*)(row);
  f.q[1] = *(const uint4*)(row + 16);
  return f.bf;
}

__device__ __forceinline__ v8f vzero8() {
  v8f r;
#pragma unroll
  for (int i = 0; i < 8; ++i) r[i] = 0.0f;
  return r;
}

#define WMMA_BF16(A, Bf, C) \
  __builtin_amdgcn_wmma_f32_16x16x32_bf16(false, (A), false, (Bf), (short)0, (C), false, false)

// ---------------------------------------------------------------------------
// CDNA5 async global->LDS copy (16 bytes / lane).  VDST = LDS byte address
// (low 32 bits of the generic shared pointer), VADDR = 64-bit global address,
// GV mode (saddr = off).  Tracked by ASYNCcnt; drain with s_wait_asynccnt 0
// before the workgroup barrier.
// ---------------------------------------------------------------------------
__device__ __forceinline__ void async_copy16(void* lds_dst, const void* gsrc) {
  unsigned lds_addr = (unsigned)(unsigned long long)lds_dst;
  asm volatile("global_load_async_to_lds_b128 %0, %1, off"
               :: "v"(lds_addr), "v"(gsrc) : "memory");
}
__device__ __forceinline__ void async_wait0() {
  asm volatile("s_wait_asynccnt 0x0" ::: "memory");
}

// ---------------------------------------------------------------------------
// f32 -> bf16 convert (grid-stride)
// ---------------------------------------------------------------------------
__global__ void cvt_f32_bf16_kernel(const float* __restrict__ in, u16* __restrict__ out, size_t n) {
  size_t i = (size_t)blockIdx.x * blockDim.x + threadIdx.x;
  size_t stride = (size_t)gridDim.x * blockDim.x;
  for (; i < n; i += stride) out[i] = f2bf(in[i]);
}

// ---------------------------------------------------------------------------
// Generic bf16 GEMM:  C[M,N](f32) = A[M,K](bf16) @ W[K,N](bf16)
// Block tile 128(M) x 128(N), 8 waves in a 2(M) x 4(N) grid; each wave owns a
// 64x32 output patch -> 8 WMMAs per K=32 step (4 A-frags x 2 B-frags).
// A tile staged via async global->LDS; W tile staged transposed.
// ---------------------------------------------------------------------------
__global__ __launch_bounds__(256) void gemm_bf16_kernel(const u16* __restrict__ A,
                                                        const u16* __restrict__ W,
                                                        float* __restrict__ C,
                                                        int M, int N, int K) {
  __shared__ u16 ldsA[128 * 32];   // [row][k]
  __shared__ u16 ldsW[128 * 32];   // transposed: [n][k]
  const int tid = threadIdx.x;
  const int w = tid >> 5, lane = tid & 31, lr = lane & 15, lh = lane >> 4;
  const int n0 = blockIdx.x * 128, m0 = blockIdx.y * 128;
  const int wm = (w & 1) * 64;     // wave M offset inside tile
  const int wn = (w >> 1) * 32;    // wave N offset inside tile

  v8f acc[4][2];
#pragma unroll
  for (int mt = 0; mt < 4; ++mt)
#pragma unroll
    for (int nt = 0; nt < 2; ++nt) acc[mt][nt] = vzero8();

  for (int k0 = 0; k0 < K; k0 += 32) {
    // A tile 128x32 : 512 16B chunks, 2 per thread, async copy to LDS
#pragma unroll
    for (int i = 0; i < 2; ++i) {
      int c = tid + i * 256;
      int row = c >> 2, cc = c & 3;
      async_copy16(&ldsA[row * 32 + cc * 8], &A[(size_t)(m0 + row) * K + k0 + cc * 8]);
    }
    // W tile 32x128 staged transposed -> ldsW[n][k] (permuting, stays manual)
#pragma unroll
    for (int i = 0; i < 2; ++i) {
      int c = tid + i * 256;
      int kk = c >> 4, nn0 = (c & 15) * 8;
      uint4 wv = *(const uint4*)&W[(size_t)(k0 + kk) * N + n0 + nn0];
      const u16* pv = (const u16*)&wv;
#pragma unroll
      for (int j = 0; j < 8; ++j) ldsW[(nn0 + j) * 32 + kk] = pv[j];
    }
    async_wait0();
    __syncthreads();

    v16bf Afr[4];
#pragma unroll
    for (int mt = 0; mt < 4; ++mt) Afr[mt] = load_frag(ldsA + (wm + mt * 16) * 32, 32);
#pragma unroll
    for (int nt = 0; nt < 2; ++nt) {
      v16bf Bf = load_frag(ldsW + (wn + nt * 16) * 32, 32);
#pragma unroll
      for (int mt = 0; mt < 4; ++mt) acc[mt][nt] = WMMA_BF16(Afr[mt], Bf, acc[mt][nt]);
    }
    __syncthreads();
  }

  // C layout: element i of v8f -> row (i + 8*lh), col lr of the 16x16 tile
#pragma unroll
  for (int mt = 0; mt < 4; ++mt)
#pragma unroll
    for (int nt = 0; nt < 2; ++nt)
#pragma unroll
      for (int i = 0; i < 8; ++i)
        C[(size_t)(m0 + wm + mt * 16 + i + 8 * lh) * N + n0 + wn + nt * 16 + lr] =
            acc[mt][nt][i];
}

// ---------------------------------------------------------------------------
// RMS norm per row, f32 in -> bf16 out
// ---------------------------------------------------------------------------
__global__ __launch_bounds__(256) void rmsnorm_bf16_kernel(const float* __restrict__ in,
                                                           const float* __restrict__ wgt,
                                                           u16* __restrict__ out, int RL) {
  __shared__ float red[256];
  const int row = blockIdx.x;
  const size_t base = (size_t)row * RL;
  float s = 0.0f;
  for (int j = threadIdx.x; j < RL; j += 256) { float v = in[base + j]; s += v * v; }
  red[threadIdx.x] = s;
  __syncthreads();
  for (int o = 128; o; o >>= 1) {
    if (threadIdx.x < o) red[threadIdx.x] += red[threadIdx.x + o];
    __syncthreads();
  }
  float inv = rsqrtf(red[0] / (float)RL + 1e-6f);
  for (int j = threadIdx.x; j < RL; j += 256)
    out[base + j] = f2bf(in[base + j] * inv * wgt[j]);
}

// ---------------------------------------------------------------------------
// Uncertainty-scaled RoPE + pack into [B,NH,T,*] bf16 Q/K/V
// ---------------------------------------------------------------------------
__global__ __launch_bounds__(256) void rope_pack_kernel(const float* __restrict__ qc,
                                                        const float* __restrict__ qr,
                                                        const float* __restrict__ kc,
                                                        const float* __restrict__ kr,
                                                        const float* __restrict__ vf,
                                                        const float* __restrict__ unc,
                                                        u16* __restrict__ Qb,
                                                        u16* __restrict__ Kb,
                                                        u16* __restrict__ Vb) {
  const int bt = blockIdx.x;
  const int b = bt >> 11, t = bt & (T_ - 1);
  float u = unc[bt];
  u = fminf(fmaxf(u, 0.0f), 1.0f);
  const float sc = 0.5f + u * 1.5f;              // MIN_SCALE + u*(MAX-MIN)
  const float LN_BASE = 13.122363377404328f;     // ln(500000)

  for (int idx = threadIdx.x; idx < NH_ * DQK_; idx += 256) {
    int h = idx / DQK_, d = idx - h * DQK_;
    size_t dsto = (((size_t)(b * NH_ + h) * T_) + t) * DQK_ + d;
    float qv, kv;
    if (d < HD_) {
      qv = qc[(size_t)bt * (NH_ * HD_) + h * HD_ + d];
      kv = kc[(size_t)bt * (NH_ * HD_) + h * HD_ + d];
    } else {
      int j = d - HD_;                 // 0..63
      int jj = j & 31;
      int second = j >> 5;
      float theta = __expf(-LN_BASE * (float)jj * 0.03125f);   // 500000^(-jj/32)
      float fr = (float)t * theta * sc;
      float c = __cosf(fr), s = __sinf(fr);
      float q1 = qr[(size_t)bt * (NH_ * R_) + h * R_ + jj];
      float q2 = qr[(size_t)bt * (NH_ * R_) + h * R_ + 32 + jj];
      float k1 = kr[(size_t)bt * (NH_ * R_) + h * R_ + jj];
      float k2 = kr[(size_t)bt * (NH_ * R_) + h * R_ + 32 + jj];
      qv = second ? (q2 * c + q1 * s) : (q1 * c - q2 * s);
      kv = second ? (k2 * c + k1 * s) : (k1 * c - k2 * s);
    }
    Qb[dsto] = f2bf(qv);
    Kb[dsto] = f2bf(kv);
  }
  for (int idx = threadIdx.x; idx < NH_ * HD_; idx += 256) {
    int h = idx >> 7, d = idx & 127;
    Vb[(((size_t)(b * NH_ + h) * T_) + t) * HD_ + d] = f2bf(vf[(size_t)bt * (NH_ * HD_) + idx]);
  }
}

// ---------------------------------------------------------------------------
// Flash-style attention.  Block = 128 Q rows of one (b,h), 8 waves (16 rows
// each).  Key loop in steps of 32: 12 QK^T WMMAs + online softmax + 8 PV
// WMMAs per step.  causal=1 restricts keys to the trailing sliding window
// [chunk_start-256, q] which exactly reproduces the reference mask.
// ---------------------------------------------------------------------------
__global__ __launch_bounds__(256) void attn_kernel(const u16* __restrict__ Qb,
                                                   const u16* __restrict__ Kb,
                                                   const u16* __restrict__ Vb,
                                                   float* __restrict__ Out, int causal) {
  __shared__ u16 ldsK[32 * DQK_];     // [key][feat]
  __shared__ u16 ldsVt[HD_ * 32];     // transposed: [col][key]
  __shared__ u16 ldsP[8 * 16 * 32];   // per-wave 16x32 P tile
  const int bh = blockIdx.y;
  const int qbase = blockIdx.x * 128;
  const int tid = threadIdx.x;
  const int w = tid >> 5, lane = tid & 31, lr = lane & 15, lh = lane >> 4;

  v16bf Aq[6];
  {
    const u16* Qrow = Qb + ((size_t)bh * T_ + qbase + w * 16) * DQK_;
#pragma unroll
    for (int kd = 0; kd < 6; ++kd) Aq[kd] = load_frag(Qrow + kd * 32, DQK_);
  }

  v8f acc[8];
  float m[8], l[8];
#pragma unroll
  for (int ct = 0; ct < 8; ++ct) acc[ct] = vzero8();
#pragma unroll
  for (int i = 0; i < 8; ++i) { m[i] = -3.0e38f; l[i] = 0.0f; }

  int lo = 0, hi = T_;
  if (causal) {
    int cs = qbase & ~255;                 // chunk start
    lo = (cs >= 256) ? cs - 256 : 0;       // window lower bound (and key>=0)
    hi = qbase + 128;                      // causal upper bound for this block
  }
  const float kscale = 0.07216878364870323f;   // 1/sqrt(192)

  for (int kt = lo; kt < hi; kt += 32) {
    // stage K tile 32x192 (row-major) via async global->LDS
#pragma unroll
    for (int i = 0; i < 3; ++i) {
      int c = tid + i * 256;
      int row = c / 24, cc = c - row * 24;
      async_copy16(&ldsK[row * DQK_ + cc * 8],
                   &Kb[((size_t)bh * T_ + kt + row) * DQK_ + cc * 8]);
    }
    // stage V tile 32x128 transposed -> ldsVt[col][key] (permuting, manual)
#pragma unroll
    for (int i = 0; i < 2; ++i) {
      int c = tid + i * 256;
      int row = c >> 4, cc = c & 15;
      uint4 v = *(const uint4*)&Vb[((size_t)bh * T_ + kt + row) * HD_ + cc * 8];
      const u16* pv = (const u16*)&v;
#pragma unroll
      for (int j = 0; j < 8; ++j) ldsVt[(cc * 8 + j) * 32 + row] = pv[j];
    }
    async_wait0();
    __syncthreads();

    // S = Q(16x192) @ K^T(192x32)  -> two 16x16 accumulators
    v8f S0 = vzero8(), S1 = vzero8();
#pragma unroll
    for (int kd = 0; kd < 6; ++kd) {
      v16bf B0 = load_frag(ldsK + kd * 32, DQK_);
      S0 = WMMA_BF16(Aq[kd], B0, S0);
      v16bf B1 = load_frag(ldsK + 16 * DQK_ + kd * 32, DQK_);
      S1 = WMMA_BF16(Aq[kd], B1, S1);
    }
#pragma unroll
    for (int i = 0; i < 8; ++i) { S0[i] *= kscale; S1[i] *= kscale; }
    if (causal) {
#pragma unroll
      for (int i = 0; i < 8; ++i) {
        int q = qbase + w * 16 + i + 8 * lh;
        if (kt + lr > q)      S0[i] = -3.0e38f;
        if (kt + 16 + lr > q) S1[i] = -3.0e38f;
      }
    }

    // online softmax: row r = i + 8*lh lives in one 16-lane half-wave
    u16* Pw = ldsP + w * 512;
#pragma unroll
    for (int i = 0; i < 8; ++i) {
      float rm = fmaxf(S0[i], S1[i]);
      rm = fmaxf(rm, __shfl_xor(rm, 1, 16));
      rm = fmaxf(rm, __shfl_xor(rm, 2, 16));
      rm = fmaxf(rm, __shfl_xor(rm, 4, 16));
      rm = fmaxf(rm, __shfl_xor(rm, 8, 16));
      float mn = fmaxf(m[i], rm);
      float corr = __expf(m[i] - mn);
      float p0 = __expf(S0[i] - mn);
      float p1 = __expf(S1[i] - mn);
      float rs = p0 + p1;
      rs += __shfl_xor(rs, 1, 16);
      rs += __shfl_xor(rs, 2, 16);
      rs += __shfl_xor(rs, 4, 16);
      rs += __shfl_xor(rs, 8, 16);
      l[i] = l[i] * corr + rs;
      m[i] = mn;
#pragma unroll
      for (int ct = 0; ct < 8; ++ct) acc[ct][i] *= corr;
      Pw[(i + 8 * lh) * 32 + lr] = f2bf(p0);
      Pw[(i + 8 * lh) * 32 + 16 + lr] = f2bf(p1);
    }
    __builtin_amdgcn_wave_barrier();   // keep DS store->load ordering tight

    // acc += P(16x32) @ V(32x128)
    v16bf Ap = load_frag(Pw, 32);
#pragma unroll
    for (int ct = 0; ct < 8; ++ct) {
      v16bf Bv = load_frag(ldsVt + ct * 16 * 32, 32);
      acc[ct] = WMMA_BF16(Ap, Bv, acc[ct]);
    }
    __syncthreads();
  }

#pragma unroll
  for (int ct = 0; ct < 8; ++ct)
#pragma unroll
    for (int i = 0; i < 8; ++i)
      Out[((size_t)bh * T_ + qbase + w * 16 + i + 8 * lh) * HD_ + ct * 16 + lr] =
          acc[ct][i] / l[i];
}

// ---------------------------------------------------------------------------
// Gate: alpha = sigmoid(x @ Wgate + b); writes ws copy + tail of d_out
// ---------------------------------------------------------------------------
__global__ __launch_bounds__(256) void gate_kernel(const float* __restrict__ x,
                                                   const float* __restrict__ Wg,
                                                   const float* __restrict__ bg,
                                                   float* __restrict__ alpha,
                                                   float* __restrict__ out_tail) {
  __shared__ float red[256];
  const int row = blockIdx.x;
  const size_t base = (size_t)row * D_;
  float s = 0.0f;
  for (int j = threadIdx.x; j < D_; j += 256) s += x[base + j] * Wg[j];
  red[threadIdx.x] = s;
  __syncthreads();
  for (int o = 128; o; o >>= 1) {
    if (threadIdx.x < o) red[threadIdx.x] += red[threadIdx.x + o];
    __syncthreads();
  }
  if (threadIdx.x == 0) {
    float a = 1.0f / (1.0f + __expf(-(red[0] + bg[0])));
    alpha[row] = a;
    out_tail[row] = a;
  }
}

// ---------------------------------------------------------------------------
// merged = a*out_bidir + (1-a)*out_ar, transposed to [b,t,NH*HD], bf16
// ---------------------------------------------------------------------------
__global__ __launch_bounds__(256) void merge_kernel(const float* __restrict__ ob,
                                                    const float* __restrict__ oa,
                                                    const float* __restrict__ alpha,
                                                    u16* __restrict__ mrg) {
  const int bt = blockIdx.x;
  const int b = bt >> 11, t = bt & (T_ - 1);
  const float a = alpha[bt];
  for (int idx = threadIdx.x; idx < NH_ * HD_; idx += 256) {
    int h = idx >> 7, d = idx & 127;
    size_t off = (((size_t)(b * NH_ + h) * T_) + t) * HD_ + d;
    mrg[(size_t)bt * (NH_ * HD_) + idx] = f2bf(a * ob[off] + (1.0f - a) * oa[off]);
  }
}

// ---------------------------------------------------------------------------
// Host launcher
// ---------------------------------------------------------------------------
extern "C" void kernel_launch(void* const* d_in, const int* in_sizes, int n_in,
                              void* d_out, int out_size, void* d_ws, size_t ws_size,
                              hipStream_t stream) {
  (void)in_sizes; (void)n_in; (void)out_size; (void)ws_size;
  const float* x        = (const float*)d_in[0];
  const float* unc      = (const float*)d_in[1];
  const float* Wq_down  = (const float*)d_in[2];
  const float* q_norm_w = (const float*)d_in[3];
  const float* Wq_up    = (const float*)d_in[4];
  const float* Wq_rope  = (const float*)d_in[5];
  const float* Wkv_down = (const float*)d_in[6];
  const float* kv_norm_w= (const float*)d_in[7];
  const float* Wk_up    = (const float*)d_in[8];
  const float* Wv_up    = (const float*)d_in[9];
  const float* Wk_rope  = (const float*)d_in[10];
  const float* Wout     = (const float*)d_in[11];
  const float* Wgate    = (const float*)d_in[12];
  const float* bgate    = (const float*)d_in[13];

  char* ws = (char*)d_ws;
  size_t off = 0;
  auto alloc = [&](size_t bytes) -> char* {
    off = (off + 255) & ~(size_t)255;
    char* p = ws + off;
    off += bytes;
    return p;
  };

  // bf16 staging
  u16* XB    = (u16*)alloc((size_t)MROWS * D_ * 2);
  u16* WqdB  = (u16*)alloc((size_t)D_ * CQ_ * 2);
  u16* WquB  = (u16*)alloc((size_t)CQ_ * (NH_ * HD_) * 2);
  u16* WqrB  = (u16*)alloc((size_t)CQ_ * (NH_ * R_) * 2);
  u16* WkvdB = (u16*)alloc((size_t)D_ * CKV_ * 2);
  u16* WkuB  = (u16*)alloc((size_t)CKV_ * (NH_ * HD_) * 2);
  u16* WvuB  = (u16*)alloc((size_t)CKV_ * (NH_ * HD_) * 2);
  u16* WkrB  = (u16*)alloc((size_t)D_ * (NH_ * R_) * 2);
  u16* WoB   = (u16*)alloc((size_t)(NH_ * HD_) * D_ * 2);
  // f32 intermediates
  float* qpre  = (float*)alloc((size_t)MROWS * CQ_ * 4);
  u16*   qlat  = (u16*)  alloc((size_t)MROWS * CQ_ * 2);
  float* kvpre = (float*)alloc((size_t)MROWS * CKV_ * 4);
  u16*   kvlat = (u16*)  alloc((size_t)MROWS * CKV_ * 2);
  float* qc    = (float*)alloc((size_t)MROWS * (NH_ * HD_) * 4);  // reused as out_bidir
  float* qr    = (float*)alloc((size_t)MROWS * (NH_ * R_) * 4);
  float* kc    = (float*)alloc((size_t)MROWS * (NH_ * HD_) * 4);  // reused as out_ar
  float* kr    = (float*)alloc((size_t)MROWS * (NH_ * R_) * 4);
  float* vf    = (float*)alloc((size_t)MROWS * (NH_ * HD_) * 4);  // reused as merged_bf
  u16*   Qb    = (u16*)alloc((size_t)B_ * NH_ * T_ * DQK_ * 2);
  u16*   Kb    = (u16*)alloc((size_t)B_ * NH_ * T_ * DQK_ * 2);
  u16*   Vb    = (u16*)alloc((size_t)B_ * NH_ * T_ * HD_ * 2);
  float* alphaB= (float*)alloc((size_t)MROWS * 4);

  float* obid = qc;          // safe: qc consumed by rope_pack before attn
  float* oar  = kc;          // safe: kc consumed by rope_pack before attn
  u16*   mrg  = (u16*)vf;    // safe: vf consumed by rope_pack before merge

  float* out_main = (float*)d_out;
  float* out_alpha = out_main + (size_t)MROWS * D_;

  // 1) bf16 conversions
  cvt_f32_bf16_kernel<<<4096, 256, 0, stream>>>(x, XB, (size_t)MROWS * D_);
  cvt_f32_bf16_kernel<<<2048, 256, 0, stream>>>(Wq_down,  WqdB,  (size_t)D_ * CQ_);
  cvt_f32_bf16_kernel<<<2048, 256, 0, stream>>>(Wq_up,    WquB,  (size_t)CQ_ * NH_ * HD_);
  cvt_f32_bf16_kernel<<<1024, 256, 0, stream>>>(Wq_rope,  WqrB,  (size_t)CQ_ * NH_ * R_);
  cvt_f32_bf16_kernel<<<1024, 256, 0, stream>>>(Wkv_down, WkvdB, (size_t)D_ * CKV_);
  cvt_f32_bf16_kernel<<<1024, 256, 0, stream>>>(Wk_up,    WkuB,  (size_t)CKV_ * NH_ * HD_);
  cvt_f32_bf16_kernel<<<1024, 256, 0, stream>>>(Wv_up,    WvuB,  (size_t)CKV_ * NH_ * HD_);
  cvt_f32_bf16_kernel<<<1024, 256, 0, stream>>>(Wk_rope,  WkrB,  (size_t)D_ * NH_ * R_);
  cvt_f32_bf16_kernel<<<2048, 256, 0, stream>>>(Wout,     WoB,   (size_t)NH_ * HD_ * D_);

  // 2) projections (all WMMA GEMMs; block tile 128x128)
  gemm_bf16_kernel<<<dim3(CQ_ / 128, MROWS / 128), 256, 0, stream>>>(XB, WqdB, qpre, MROWS, CQ_, D_);
  rmsnorm_bf16_kernel<<<MROWS, 256, 0, stream>>>(qpre, q_norm_w, qlat, CQ_);
  gemm_bf16_kernel<<<dim3((NH_*HD_) / 128, MROWS / 128), 256, 0, stream>>>(qlat, WquB, qc, MROWS, NH_*HD_, CQ_);
  gemm_bf16_kernel<<<dim3((NH_*R_) / 128, MROWS / 128), 256, 0, stream>>>(qlat, WqrB, qr, MROWS, NH_*R_, CQ_);
  gemm_bf16_kernel<<<dim3(CKV_ / 128, MROWS / 128), 256, 0, stream>>>(XB, WkvdB, kvpre, MROWS, CKV_, D_);
  rmsnorm_bf16_kernel<<<MROWS, 256, 0, stream>>>(kvpre, kv_norm_w, kvlat, CKV_);
  gemm_bf16_kernel<<<dim3((NH_*HD_) / 128, MROWS / 128), 256, 0, stream>>>(kvlat, WkuB, kc, MROWS, NH_*HD_, CKV_);
  gemm_bf16_kernel<<<dim3((NH_*HD_) / 128, MROWS / 128), 256, 0, stream>>>(kvlat, WvuB, vf, MROWS, NH_*HD_, CKV_);
  gemm_bf16_kernel<<<dim3((NH_*R_) / 128, MROWS / 128), 256, 0, stream>>>(XB, WkrB, kr, MROWS, NH_*R_, D_);

  // 3) RoPE + head packing
  rope_pack_kernel<<<MROWS, 256, 0, stream>>>(qc, qr, kc, kr, vf, unc, Qb, Kb, Vb);

  // 4) attention (bidirectional full + causal sliding window)
  attn_kernel<<<dim3(T_ / 128, B_ * NH_), 256, 0, stream>>>(Qb, Kb, Vb, obid, 0);
  attn_kernel<<<dim3(T_ / 128, B_ * NH_), 256, 0, stream>>>(Qb, Kb, Vb, oar, 1);

  // 5) gate + merge
  gate_kernel<<<MROWS, 256, 0, stream>>>(x, Wgate, bgate, alphaB, out_alpha);
  merge_kernel<<<MROWS, 256, 0, stream>>>(obid, oar, alphaB, mrg);

  // 6) output projection straight into d_out
  gemm_bf16_kernel<<<dim3(D_ / 128, MROWS / 128), 256, 0, stream>>>(mrg, WoB, out_main, MROWS, D_, NH_ * HD_);
}
